// WindowedMultiheadAttention_27994596835728
// MI455X (gfx1250) — matrix operand order repaired and checked by reading, not compiled, not run on hardware
//
#include <hip/hip_runtime.h>

// ---------------------------------------------------------------------------
// CDNA5 (gfx1250, wave32) windowed multi-head attention.
// bf16 WMMA 16x16x32 with fp32 accumulation everywhere; analytic window mask
// (mask tensor input is never read); flash-style online softmax; no NxN
// score materialization.
// ---------------------------------------------------------------------------

typedef __bf16 bf16;
typedef bf16  v16bf __attribute__((ext_vector_type(16)));
typedef float v8f   __attribute__((ext_vector_type(8)));

static constexpr int   kBatch = 4;
static constexpr int   kN     = 2304;      // 48*48 tokens
static constexpr int   kE     = 128;
static constexpr int   kNH    = 8;
static constexpr int   kHD    = 16;
static constexpr int   kGW    = 48;        // grid width
static constexpr int   kMT    = (kBatch * kN) / 16;  // 576 row tiles over B*N
static constexpr int   kQT    = kN / 16;             // 144 query tiles per batch
static constexpr float kScale = 0.08838834764831843f; // 128^-0.5
static constexpr size_t kSZ   = (size_t)kBatch * kNH * kN * kHD; // 1179648 elems

#define WMMA_BF16(a, b, c) \
  __builtin_amdgcn_wmma_f32_16x16x32_bf16(false, (a), false, (b), (short)0, (c), false, false)

// ---- fragment builders (ISA 7.12.2 layouts, wave32) -----------------------
// A 16x32 bf16: lane<16 row=lane, K elems {0..7,16..23}; lane>=16 K+=8.
static __device__ __forceinline__ v16bf frag_a_f32(const float* rowp, int half) {
  v16bf a;
#pragma unroll
  for (int e = 0; e < 8; ++e) {
    a[e]     = (bf16)rowp[half * 8 + e];
    a[e + 8] = (bf16)rowp[16 + half * 8 + e];
  }
  return a;
}
static __device__ __forceinline__ v16bf frag_a_bf16(const bf16* rowp, int half) {
  v16bf a;
#pragma unroll
  for (int e = 0; e < 8; ++e) {
    a[e]     = rowp[half * 8 + e];
    a[e + 8] = rowp[16 + half * 8 + e];
  }
  return a;
}
// A with only K=0..15 valid (head_dim 16 zero-padded to 32).
static __device__ __forceinline__ v16bf frag_a_k16(const bf16* rowp, int half) {
  v16bf a;
#pragma unroll
  for (int e = 0; e < 8; ++e) {
    a[e]     = rowp[half * 8 + e];
    a[e + 8] = (bf16)0.0f;
  }
  return a;
}
// B 32x16 bf16: col = lane%16; lane<16 K elems {0..7,16..23}; lane>=16 K+=8.
static __device__ __forceinline__ v16bf frag_b_f32(const float* rowp, int half) {
  v16bf b;
#pragma unroll
  for (int e = 0; e < 16; ++e) b[e] = (bf16)rowp[e < 8 ? half * 8 + e : 8 + half * 8 + e];
  return b;
}
// B with only K=0..15 valid (elements 8..15 of each lane hold K>=16: zeros).
static __device__ __forceinline__ v16bf frag_b_k16(const bf16* rowp, int half) {
  v16bf b;
#pragma unroll
  for (int e = 0; e < 16; ++e) b[e] = (e >= 8) ? (bf16)0.0f : rowp[half * 8 + e];
  return b;
}

// row reductions across the 16 lanes holding one C-matrix row
static __device__ __forceinline__ float rowmax16(float v) {
#pragma unroll
  for (int off = 1; off < 16; off <<= 1) v = fmaxf(v, __shfl_xor(v, off, 16));
  return v;
}
static __device__ __forceinline__ float rowsum16(float v) {
#pragma unroll
  for (int off = 1; off < 16; off <<= 1) v += __shfl_xor(v, off, 16);
  return v;
}

// ---------------------------------------------------------------------------
// Kernel 1: qkv = x @ qkv_w^T + qkv_b  (q pre-scaled by E^-0.5)
// grid (576, 3), 256 threads (8 waves). Wave handles one 16x16 output tile.
// Q,K -> bf16 [B,H,N,16]; V -> bf16 [B,H,16,N] (transposed for AV B-frags).
// ---------------------------------------------------------------------------
__global__ __launch_bounds__(256) void qkv_kernel(
    const float* __restrict__ x, const float* __restrict__ qkv_w,
    const float* __restrict__ qkv_b, bf16* __restrict__ qws,
    bf16* __restrict__ kws, bf16* __restrict__ vws) {
  const int lane = threadIdx.x & 31, wave = threadIdx.x >> 5;
  const int half = lane >> 4, l16 = lane & 15;
  const int bn0 = blockIdx.x * 16;               // row tile over B*N
  const int jt  = blockIdx.y * 8 + wave;         // 0..23 column tiles of 384
  const int j0  = jt * 16;

  const float* xrow = x + (size_t)(bn0 + l16) * kE;
  const float* wrow = qkv_w + (size_t)(j0 + l16) * kE;
  __builtin_prefetch(xrow, 0, 1);

  v8f c = {0.f, 0.f, 0.f, 0.f, 0.f, 0.f, 0.f, 0.f};
#pragma unroll
  for (int k0 = 0; k0 < kE; k0 += 32) {
    v16bf a  = frag_a_f32(xrow + k0, half);
    v16bf bw = frag_b_f32(wrow + k0, half);
    c = WMMA_BF16(a, bw, c);
  }

  const int j     = j0 + l16;
  const float bia = qkv_b[j];
  const int which = j0 >> 7;             // 0=q 1=k 2=v (tile never straddles)
  const int head  = (j0 & 127) >> 4;
  const int b     = bn0 / kN;
  const int nn0   = bn0 % kN;

#pragma unroll
  for (int r = 0; r < 8; ++r) {
    const int row = r + half * 8;
    const int n   = nn0 + row;
    float v = c[r] + bia;
    if (which == 0) v *= kScale;
    if (which == 2) {
      vws[((size_t)(b * kNH + head) * kHD + l16) * kN + n] = (bf16)v;
    } else {
      bf16* dst = (which == 0) ? qws : kws;
      dst[((size_t)(b * kNH + head) * kN + n) * kHD + l16] = (bf16)v;
    }
  }
}

// ---------------------------------------------------------------------------
// Kernel 2: windowed attention, flash-style. grid = B*144 blocks; wave = head.
// ---------------------------------------------------------------------------
__global__ __launch_bounds__(256) void attn_kernel(
    const bf16* __restrict__ qws, const bf16* __restrict__ kws,
    const bf16* __restrict__ vws, bf16* __restrict__ aws) {
  __shared__ bf16 Pl[8][16][16];   // per-wave P tile staging (C->A re-layout)

  const int lane = threadIdx.x & 31, wave = threadIdx.x >> 5;
  const int half = lane >> 4, l16 = lane & 15;
  const int b  = blockIdx.x / kQT;
  const int tq = blockIdx.x % kQT;
  const int n0 = tq * 16;
  const int hq = n0 / kGW, w0 = n0 % kGW;   // tile of 16 queries, same grid row
  const int head = wave;
  const size_t bh = (size_t)(b * kNH + head);

  // Q fragment (row = query = l16, K = head-dim padded 16->32)
  const v16bf aq = frag_a_k16(qws + (bh * kN + n0 + l16) * kHD, half);

  v8f o = {0.f, 0.f, 0.f, 0.f, 0.f, 0.f, 0.f, 0.f};
  float m[8], l[8];
#pragma unroll
  for (int r = 0; r < 8; ++r) { m[r] = -1e30f; l[r] = 0.f; }

  const bf16* vrow = vws + (bh * kHD + l16) * kN;  // V^T row for this d=l16

  for (int hk = hq - 3; hk <= hq + 3; ++hk) {
    if (hk < 0 || hk >= kGW) continue;
    for (int kt = 0; kt < 3; ++kt) {
      // skip key tiles with no |dw|<=5 overlap against query cols [w0,w0+15]
      if (kt * 16 + 15 < w0 - 5 || kt * 16 > w0 + 20) continue;
      const int kb = hk * kGW + kt * 16;

      // S = Q K^T  (one bf16 WMMA; contraction = head-dim, zero padded)
      const v16bf bk = frag_b_k16(kws + (bh * kN + kb + l16) * kHD, half);
      v8f s = {0.f, 0.f, 0.f, 0.f, 0.f, 0.f, 0.f, 0.f};
      s = WMMA_BF16(aq, bk, s);

      // analytic window mask + online softmax update
      const int wk = kt * 16 + l16;
      float p[8];
#pragma unroll
      for (int r = 0; r < 8; ++r) {
        const int row = r + half * 8;
        const int wq  = w0 + row;
        const bool ok = (wq - wk <= 5) && (wk - wq <= 5);
        const float sv = ok ? s[r] : -1e30f;
        const float mnew  = fmaxf(m[r], rowmax16(sv));
        const float alpha = __expf(m[r] - mnew);
        const float pv    = ok ? __expf(sv - mnew) : 0.0f;
        l[r] = l[r] * alpha + rowsum16(pv);
        m[r] = mnew;
        o[r] *= alpha;
        p[r] = pv;
      }

      // P: C-layout -> LDS -> A-layout fragment
#pragma unroll
      for (int r = 0; r < 8; ++r)
        Pl[wave][r + half * 8][l16] = (bf16)p[r];
      __asm__ volatile("s_wait_dscnt 0" ::: "memory");
      __builtin_amdgcn_wave_barrier();

      v16bf ap;
#pragma unroll
      for (int e = 0; e < 8; ++e) {
        ap[e]     = Pl[wave][l16][half * 8 + e];
        ap[e + 8] = (bf16)0.0f;
      }
      const v16bf bv = frag_b_k16(vrow + kb, half);  // contiguous: V transposed
      o = WMMA_BF16(ap, bv, o);
      __builtin_amdgcn_wave_barrier();
    }
  }

  // normalize and store per-head output interleaved as [B, N, 128] bf16
#pragma unroll
  for (int r = 0; r < 8; ++r) {
    const int row = r + half * 8;
    const float val = o[r] / l[r];
    aws[((size_t)b * kN + n0 + row) * kE + head * kHD + l16] = (bf16)val;
  }
}

// ---------------------------------------------------------------------------
// Kernel 3: out = attn @ proj_w^T + proj_b. grid 576 blocks; wave = j tile.
// ---------------------------------------------------------------------------
__global__ __launch_bounds__(256) void proj_kernel(
    const bf16* __restrict__ aws, const float* __restrict__ proj_w,
    const float* __restrict__ proj_b, float* __restrict__ out) {
  const int lane = threadIdx.x & 31, wave = threadIdx.x >> 5;
  const int half = lane >> 4, l16 = lane & 15;
  const int bn0 = blockIdx.x * 16;
  const int j0  = wave * 16;

  const bf16*  arow = aws + (size_t)(bn0 + l16) * kE;
  const float* wrow = proj_w + (size_t)(j0 + l16) * kE;

  v8f c = {0.f, 0.f, 0.f, 0.f, 0.f, 0.f, 0.f, 0.f};
#pragma unroll
  for (int k0 = 0; k0 < kE; k0 += 32) {
    v16bf a  = frag_a_bf16(arow + k0, half);
    v16bf bw = frag_b_f32(wrow + k0, half);
    c = WMMA_BF16(a, bw, c);
  }

  const int j = j0 + l16;
  const float bia = proj_b[j];
#pragma unroll
  for (int r = 0; r < 8; ++r)
    out[(size_t)(bn0 + r + half * 8) * kE + j] = c[r] + bia;
}

// ---------------------------------------------------------------------------
extern "C" void kernel_launch(void* const* d_in, const int* in_sizes, int n_in,
                              void* d_out, int out_size, void* d_ws, size_t ws_size,
                              hipStream_t stream) {
  (void)in_sizes; (void)n_in; (void)out_size; (void)ws_size;
  const float* x      = (const float*)d_in[0];
  const float* qkv_w  = (const float*)d_in[1];
  const float* qkv_b  = (const float*)d_in[2];
  const float* proj_w = (const float*)d_in[3];
  const float* proj_b = (const float*)d_in[4];
  // d_in[5] (dense [1,1,N,N] mask, 21 MB) intentionally unused: mask is analytic.

  bf16* ws  = (bf16*)d_ws;
  bf16* qws = ws;            // [B,H,N,16]  bf16
  bf16* kws = ws + kSZ;      // [B,H,N,16]  bf16
  bf16* vws = ws + 2 * kSZ;  // [B,H,16,N]  bf16 (transposed)
  bf16* aws = ws + 3 * kSZ;  // [B,N,128]   bf16 attention output

  qkv_kernel<<<dim3(kMT, 3), 256, 0, stream>>>(x, qkv_w, qkv_b, qws, kws, vws);
  attn_kernel<<<dim3(kBatch * kQT), 256, 0, stream>>>(qws, kws, vws, aws);
  proj_kernel<<<dim3(kMT), 256, 0, stream>>>(aws, proj_w, proj_b, (float*)d_out);
}
